// KimiMLAAttention_74371653698286
// MI455X (gfx1250) — matrix operand order repaired and checked by reading, not compiled, not run on hardware
//
#include <hip/hip_runtime.h>

// ---------------------------------------------------------------------------
// MLA attention for gfx1250 (MI455X): all GEMM stages via v_wmma_f32_16x16x32_bf16
// Attention K-tiles double-buffered in LDS via global_load_async_to_lds_b128.
// Wave-uniform control scalarized via readfirstlane.
// ---------------------------------------------------------------------------

typedef __bf16 bf16;
typedef bf16  bf16x4  __attribute__((ext_vector_type(4)));
typedef bf16  bf16x8  __attribute__((ext_vector_type(8)));
typedef bf16  bf16x16 __attribute__((ext_vector_type(16)));
typedef float floatx8 __attribute__((ext_vector_type(8)));

__device__ __forceinline__ bf16x16 frag_join(const bf16* lo, const bf16* hi) {
  union { bf16x16 f; bf16x8 h[2]; } u;
  u.h[0] = *(const bf16x8*)lo;   // b128 load
  u.h[1] = *(const bf16x8*)hi;   // b128 load
  return u.f;
}
__device__ __forceinline__ bf16x16 frag16(const bf16* p) { return frag_join(p, p + 8); }

__device__ __forceinline__ floatx8 wmma_bf16(bf16x16 a, bf16x16 b, floatx8 c) {
  return __builtin_amdgcn_wmma_f32_16x16x32_bf16(false, a, false, b, (short)0, c, false, false);
}

// async copy: each lane moves 16B global -> LDS (tracked by ASYNCcnt)
__device__ __forceinline__ void async_copy_b128(const void* gsrc, void* ldst) {
  const unsigned long long ga = (unsigned long long)(uintptr_t)gsrc;
  const unsigned la = (unsigned)(uintptr_t)ldst;   // LDS aperture: low 32 bits = LDS addr
  asm volatile("global_load_async_to_lds_b128 %0, %1, off" :: "v"(la), "v"(ga) : "memory");
}
__device__ __forceinline__ void wait_async(int n) {  // imm-selected variants
  if (n == 0)  asm volatile("s_wait_asynccnt 0x0"  ::: "memory");
  else         asm volatile("s_wait_asynccnt 0x18" ::: "memory");   // 24 outstanding
}
__device__ __forceinline__ void wait_ds0() {
  asm volatile("s_wait_dscnt 0x0" ::: "memory");
}

// ---------------------------------------------------------------------------
// fp32 -> bf16 conversion (vectorized x4)
// ---------------------------------------------------------------------------
__global__ void f32_to_bf16_kernel(const float* __restrict__ in, bf16* __restrict__ out, int n4) {
  const int i = blockIdx.x * blockDim.x + threadIdx.x;
  if (i >= n4) return;
  const float4 v = ((const float4*)in)[i];
  bf16x4 o;
  o[0] = (bf16)v.x; o[1] = (bf16)v.y; o[2] = (bf16)v.z; o[3] = (bf16)v.w;
  ((bf16x4*)out)[i] = o;
}

// ---------------------------------------------------------------------------
// Generic C = A(MxK,bf16,row-major) * W(NxK,bf16,row-major)^T
// One wave computes a 16 x (16*NT) tile (A-fragment reused across NT WMMAs).
// The 4 waves of a block share tn (same B fragments -> WGP$ hits) and take
// adjacent tm.
// MODE 0: fp32 row-major out (Cf)
// MODE 1: bf16 row-major out (Cb)
// MODE 2: kv_b split writer: col = h*256 + j; j<128 -> kpass[tok,h,j] (Cb),
//         j>=128 -> vT[b,h,j-128,s] (Cb2)  (pre-transposed V for attention)
// ---------------------------------------------------------------------------
template<int MODE, int NT>
__global__ __launch_bounds__(128, 1) void wmma_gemm_nt(
    const bf16* __restrict__ A, const bf16* __restrict__ W,
    float* __restrict__ Cf, bf16* __restrict__ Cb, bf16* __restrict__ Cb2,
    int M, int N, int K) {
  const int lane = threadIdx.x & 31;
  const int wave = __builtin_amdgcn_readfirstlane(
      (blockIdx.x * blockDim.x + threadIdx.x) >> 5);      // wave-uniform -> SGPR
  const int tilesM = M >> 4;
  const int tm = wave % tilesM;                 // adjacent M tiles within a block
  const int tn = (wave / tilesM) * (16 * NT);   // same N tile for all 4 waves of a block
  const int m0 = tm << 4;
  if (tn >= N) return;
  const int hl = lane >> 4, l16 = lane & 15;

  const bf16* arow  = A + (size_t)(m0 + l16) * K;
  const bf16* wrow0 = W + (size_t)(tn + l16) * K;

  floatx8 acc[NT] = {};
  for (int k0 = 0; k0 < K; k0 += 32) {
    __builtin_prefetch(arow + k0 + 256, 0, 0);     // global_prefetch_b8 ahead on stream
    const bf16x16 af = frag16(arow + k0 + hl * 16);
#pragma unroll
    for (int j = 0; j < NT; ++j) {
      const bf16x16 bfv = frag16(wrow0 + (size_t)(j * 16) * K + k0 + hl * 16);
      acc[j] = wmma_bf16(af, bfv, acc[j]);
    }
  }

#pragma unroll
  for (int j = 0; j < NT; ++j) {
#pragma unroll
    for (int r = 0; r < 8; ++r) {
      const int row = m0 + r + hl * 8;        // C layout: VGPR r + half -> M
      const int col = tn + j * 16 + l16;      // lane%16 -> N
      const float v = acc[j][r];
      if (MODE == 0) {
        Cf[(size_t)row * N + col] = v;
      } else if (MODE == 1) {
        Cb[(size_t)row * N + col] = (bf16)v;
      } else {
        const int hh = col >> 8, jc = col & 255;
        if (jc < 128) {
          Cb[(size_t)row * 2048 + hh * 128 + jc] = (bf16)v;       // kpass[tok, h, j]
        } else {
          const int bb = row >> 11, ss = row & 2047;              // S = 2048
          Cb2[(((size_t)(bb * 16 + hh)) * 128 + (jc - 128)) * 2048 + ss] = (bf16)v;
        }
      }
    }
  }
}

// ---------------------------------------------------------------------------
// RMSNorm over latent 512 + bf16 split of rope 64; one wave per token row.
// ---------------------------------------------------------------------------
__global__ void rmsnorm_split_kernel(const float* __restrict__ comp, const float* __restrict__ w,
                                     bf16* __restrict__ ckv, bf16* __restrict__ krot) {
  const int lane = threadIdx.x & 31;
  const int row  = __builtin_amdgcn_readfirstlane(
      (blockIdx.x * blockDim.x + threadIdx.x) >> 5);
  const float* x = comp + (size_t)row * 576;
  float v[16];
  float ss = 0.f;
#pragma unroll
  for (int i = 0; i < 16; ++i) { v[i] = x[lane + i * 32]; ss += v[i] * v[i]; }
#pragma unroll
  for (int off = 16; off; off >>= 1) ss += __shfl_xor(ss, off, 32);
  const float sc = rsqrtf(ss * (1.0f / 512.0f) + 1e-6f);
  bf16* o = ckv + (size_t)row * 512;
#pragma unroll
  for (int i = 0; i < 16; ++i) o[lane + i * 32] = (bf16)(w[lane + i * 32] * v[i] * sc);
  krot[(size_t)row * 64 + lane * 2]     = (bf16)x[512 + lane * 2];
  krot[(size_t)row * 64 + lane * 2 + 1] = (bf16)x[512 + lane * 2 + 1];
}

// ---------------------------------------------------------------------------
// Flash attention pieces
// ---------------------------------------------------------------------------
namespace attn_cfg {
constexpr int Sx = 2048, Hx = 16;
constexpr int KP_STRIDE = 136, KR_STRIDE = 72;   // padded rows -> conflict-free b128 reads
}

// Stage a 32-key (nope 128 + rope 64) tile into LDS; 24 async b128 issues.
__device__ __forceinline__ void stage_k_tile(const bf16* __restrict__ kpass,
                                             const bf16* __restrict__ krot,
                                             size_t tokBase, int h, int k0,
                                             bf16* Kp, bf16* Kr, int lane) {
  using namespace attn_cfg;
  const int cL = lane & 15, rL = lane >> 4;     // 2 keys / issue, 16B per lane
#pragma unroll
  for (int i = 0; i < 16; ++i) {
    const int kl = 2 * i + rL;
    async_copy_b128(kpass + (tokBase + k0 + kl) * 2048 + h * 128 + cL * 8,
                    Kp + kl * KP_STRIDE + cL * 8);
  }
  const int c8 = lane & 7, r8 = lane >> 3;      // 4 keys / issue
#pragma unroll
  for (int i = 0; i < 8; ++i) {
    const int kl = 4 * i + r8;
    async_copy_b128(krot + (tokBase + k0 + kl) * 64 + c8 * 8,
                    Kr + kl * KR_STRIDE + c8 * 8);
  }
}

// Online softmax update for one 16x32 score block; masking compiled out when
// the block is strictly below the causal diagonal.
template<bool MASKED>
__device__ __forceinline__ void softmax_update(floatx8 (&sc)[2], floatx8 (&O)[8],
                                               float (&mrow)[8], float (&lrow)[8],
                                               bf16* P, int q0, int k0,
                                               int hl, int l16, float scaling) {
#pragma unroll
  for (int r = 0; r < 8; ++r) {
    const int qi = q0 + r + hl * 8;
    float v0 = sc[0][r] * scaling;
    float v1 = sc[1][r] * scaling;
    if (MASKED) {
      if (k0 + l16 > qi)      v0 = -1.0e9f;
      if (k0 + 16 + l16 > qi) v1 = -1.0e9f;
    }
    float mloc = fmaxf(v0, v1);
#pragma unroll
    for (int off = 8; off; off >>= 1) mloc = fmaxf(mloc, __shfl_xor(mloc, off, 32));
    const float mnew  = fmaxf(mrow[r], mloc);
    const float alpha = __expf(mrow[r] - mnew);
    const float p0 = __expf(v0 - mnew), p1 = __expf(v1 - mnew);
    float rs = p0 + p1;
#pragma unroll
    for (int off = 8; off; off >>= 1) rs += __shfl_xor(rs, off, 32);
    lrow[r] = lrow[r] * alpha + rs;
    mrow[r] = mnew;
#pragma unroll
    for (int j = 0; j < 8; ++j) O[j][r] *= alpha;
    const int M = r + hl * 8;
    P[M * 32 + l16]      = (bf16)p0;
    P[M * 32 + 16 + l16] = (bf16)p1;
  }
}

// ---------------------------------------------------------------------------
// Flash attention: one wave per (b, h, 16-query tile); 32-key blocks,
// K tile double-buffered in LDS (async copies overlap the previous block's
// WMMAs); V read direct from pre-transposed vT (contiguous along key).
// ---------------------------------------------------------------------------
__global__ __launch_bounds__(128, 1) void mla_attn_kernel(
    const bf16* __restrict__ q, const bf16* __restrict__ kpass,
    const bf16* __restrict__ krot, const bf16* __restrict__ vT,
    bf16* __restrict__ attn) {
  using namespace attn_cfg;
  const int lane = threadIdx.x & 31;
  const int wib  = __builtin_amdgcn_readfirstlane(threadIdx.x >> 5);  // wave-uniform
  const int waveId = blockIdx.x * 4 + wib;                            // SGPR value
  const int qt = waveId & 127;       // S/16
  const int bh = waveId >> 7;        // 0..31
  const int h  = bh & 15;
  const int b  = bh >> 4;
  const int q0 = qt << 4;
  const int hl = lane >> 4, l16 = lane & 15;

  __shared__ alignas(16) bf16 ldsKp[4][2][32 * KP_STRIDE];
  __shared__ alignas(16) bf16 ldsKr[4][2][32 * KR_STRIDE];
  __shared__ alignas(16) bf16 ldsP[4][16 * 32];
  bf16* KpB[2] = { &ldsKp[wib][0][0], &ldsKp[wib][1][0] };
  bf16* KrB[2] = { &ldsKr[wib][0][0], &ldsKr[wib][1][0] };
  bf16* P = ldsP[wib];

  // Q fragments: 16 rows x 192, chunks of 32 along qdim (held in VGPRs)
  const bf16* qrow = q + ((size_t)(b * Sx + q0 + l16)) * 3072 + h * 192;
  bf16x16 qfrag[6];
#pragma unroll
  for (int c = 0; c < 6; ++c)
    qfrag[c] = frag16(qrow + c * 32 + hl * 16);

  floatx8 O[8] = {};
  float mrow[8], lrow[8];
#pragma unroll
  for (int r = 0; r < 8; ++r) { mrow[r] = -3.0e38f; lrow[r] = 0.f; }

  const size_t tokBase = (size_t)b * Sx;
  const bf16* vbase = vT + ((size_t)(b * Hx + h) * 128) * Sx;
  const float scaling = 0.07216878364870322f;   // 192^-0.5

  const int nkb = (q0 + 47) >> 5;               // ceil((q0+16)/32), SGPR

  // prologue: stage block 0 into buffer 0
  stage_k_tile(kpass, krot, tokBase, h, 0, KpB[0], KrB[0], lane);

  for (int kb = 0; kb < nkb; ++kb) {
    const int k0 = kb << 5;
    const int cur = kb & 1;

    if (kb + 1 < nkb) {                         // scalar branch (s_cbranch)
      wait_ds0();                               // drain reads of the buffer we overwrite
      stage_k_tile(kpass, krot, tokBase, h, k0 + 32, KpB[cur ^ 1], KrB[cur ^ 1], lane);
      wait_async(24);                           // block kb's 24 copies retired (in-order)
    } else {
      wait_async(0);
    }

    // --- scores: two 16x16 tiles over 32 keys, K-fragments from LDS ---
    floatx8 sc[2] = {};
#pragma unroll
    for (int t = 0; t < 2; ++t) {
      const bf16* kp = KpB[cur] + (t * 16 + l16) * KP_STRIDE;   // B-frag lane -> key
      const bf16* kr = KrB[cur] + (t * 16 + l16) * KR_STRIDE;
#pragma unroll
      for (int c = 0; c < 4; ++c)
        sc[t] = wmma_bf16(qfrag[c], frag16(kp + c * 32 + hl * 16), sc[t]);
#pragma unroll
      for (int c = 0; c < 2; ++c)
        sc[t] = wmma_bf16(qfrag[4 + c], frag16(kr + c * 32 + hl * 16), sc[t]);
    }

    // --- online softmax; mask only the diagonal blocks (scalar branch) ---
    if (k0 + 31 > q0)
      softmax_update<true >(sc, O, mrow, lrow, P, q0, k0, hl, l16, scaling);
    else
      softmax_update<false>(sc, O, mrow, lrow, P, q0, k0, hl, l16, scaling);

    // --- O += P(16x32) * V(32x128): A from LDS, B contiguous from vT ---
    const bf16x16 pfrag = frag16(P + l16 * 32 + hl * 16);
#pragma unroll
    for (int j = 0; j < 8; ++j) {
      const bf16x16 vf = frag16(vbase + (size_t)(j * 16 + l16) * Sx + k0 + hl * 16);
      O[j] = wmma_bf16(pfrag, vf, O[j]);
    }
  }

  // epilogue: normalize, store bf16 attn output (tok, h*128+d)
#pragma unroll
  for (int r = 0; r < 8; ++r) {
    const int M = r + hl * 8;
    const float inv = __builtin_amdgcn_rcpf(lrow[r]);   // v_rcp_f32
    bf16* orow = attn + (tokBase + q0 + M) * 2048 + h * 128;
#pragma unroll
    for (int j = 0; j < 8; ++j) orow[j * 16 + l16] = (bf16)(O[j][r] * inv);
  }
}

// ---------------------------------------------------------------------------
// Host-side orchestration
// ---------------------------------------------------------------------------
static inline size_t align256(size_t x) { return (x + 255) & ~(size_t)255; }

extern "C" void kernel_launch(void* const* d_in, const int* in_sizes, int n_in,
                              void* d_out, int out_size, void* d_ws, size_t ws_size,
                              hipStream_t stream) {
  (void)in_sizes; (void)n_in; (void)out_size; (void)ws_size;
  const float* hidden = (const float*)d_in[0];   // (2,2048,2048)
  const float* qw     = (const float*)d_in[1];   // (3072,2048)
  const float* kvaw   = (const float*)d_in[2];   // (576,2048)
  const float* lnw    = (const float*)d_in[3];   // (512)
  const float* kvbw   = (const float*)d_in[4];   // (4096,512)
  const float* ow     = (const float*)d_in[5];   // (2048,2048)
  float* out = (float*)d_out;                    // (2,2048,2048)

  const int TOK = 2 * 2048;                      // 4096 tokens
  char* base = (char*)d_ws;
  size_t off = 0;
  auto carve = [&](size_t bytes) -> char* { char* r = base + off; off = align256(off + bytes); return r; };

  bf16*  h_bf    = (bf16*)carve((size_t)TOK * 2048 * 2);
  bf16*  qw_bf   = (bf16*)carve((size_t)3072 * 2048 * 2);
  bf16*  kvaw_bf = (bf16*)carve((size_t)576 * 2048 * 2);
  bf16*  kvbw_bf = (bf16*)carve((size_t)4096 * 512 * 2);
  bf16*  ow_bf   = (bf16*)carve((size_t)2048 * 2048 * 2);
  bf16*  q_bf    = (bf16*)carve((size_t)TOK * 3072 * 2);
  float* comp    = (float*)carve((size_t)TOK * 576 * 4);
  bf16*  ckv     = (bf16*)carve((size_t)TOK * 512 * 2);
  bf16*  krot    = (bf16*)carve((size_t)TOK * 64 * 2);
  bf16*  kpass   = (bf16*)carve((size_t)TOK * 2048 * 2);
  bf16*  vT      = (bf16*)carve((size_t)TOK * 2048 * 2);
  bf16*  attn    = (bf16*)carve((size_t)TOK * 2048 * 2);

  auto conv = [&](const float* src, bf16* dst, size_t n) {
    const int n4 = (int)(n >> 2);
    f32_to_bf16_kernel<<<(n4 + 255) / 256, 256, 0, stream>>>(src, dst, n4);
  };
  conv(hidden, h_bf,    (size_t)TOK * 2048);
  conv(qw,     qw_bf,   (size_t)3072 * 2048);
  conv(kvaw,   kvaw_bf, (size_t)576 * 2048);
  conv(kvbw,   kvbw_bf, (size_t)4096 * 512);
  conv(ow,     ow_bf,   (size_t)2048 * 2048);

  // q = hidden @ q_proj_w^T  -> bf16 (tok, 3072); 16x128 tiles
  { const int waves = (TOK / 16) * (3072 / 128);
    wmma_gemm_nt<1, 8><<<waves / 4, 128, 0, stream>>>(h_bf, qw_bf, nullptr, q_bf, nullptr, TOK, 3072, 2048); }
  // compressed = hidden @ kv_a_proj_w^T -> fp32 (tok, 576); 16x64 tiles
  { const int waves = (TOK / 16) * (576 / 64);
    wmma_gemm_nt<0, 4><<<waves / 4, 128, 0, stream>>>(h_bf, kvaw_bf, comp, nullptr, nullptr, TOK, 576, 2048); }
  // rmsnorm latent -> ckv bf16; split rope -> krot bf16
  rmsnorm_split_kernel<<<TOK / 4, 128, 0, stream>>>(comp, lnw, ckv, krot);
  // kv = ckv @ kv_b_proj_w^T, split: k_pass row-major + V pre-transposed; 16x128 tiles
  { const int waves = (TOK / 16) * (4096 / 128);
    wmma_gemm_nt<2, 8><<<waves / 4, 128, 0, stream>>>(ckv, kvbw_bf, nullptr, kpass, vT, TOK, 4096, 512); }
  // flash attention
  mla_attn_kernel<<<(2 * 16 * 128) / 4, 128, 0, stream>>>(q_bf, kpass, krot, vT, attn);
  // out = attn @ o_proj_w^T -> fp32; 16x128 tiles
  { const int waves = (TOK / 16) * (2048 / 128);
    wmma_gemm_nt<0, 8><<<waves / 4, 128, 0, stream>>>(attn, ow_bf, out, nullptr, nullptr, TOK, 2048, 2048); }
}
